// PositiveWaveFunction_50422916055570
// MI455X (gfx1250) — matrix operand order repaired and checked by reading, not compiled, not run on hardware
//
#include <hip/hip_runtime.h>

// CDNA5 / gfx1250, wave32. Sequential RNN wavefunction.
// 256 blocks x 4 waves (128 thr). Each wave owns 2 of 8 N-tiles of the
// K-augmented (K=160) recurrent WMMA GEMM with its B fragments held in
// registers; h (16x160 f16, K-permuted) is double-buffered in LDS with one
// __syncthreads per step. Logit/softmax/masking duty rotates across waves.
// Bias + b_lin are folded into the GEMM via augment features (x0, x1, 1).
// A chunks preloaded into distinct regs (batched ds loads), split partial
// accumulators (WMMA dep depth <=3), spins software-pipelined + prefetched.

typedef __attribute__((ext_vector_type(16))) _Float16 v16h;
typedef __attribute__((ext_vector_type(8)))  _Float16 v8h;
typedef __attribute__((ext_vector_type(8)))  float    v8f;

#define S_SPINS 1024
#define BATCH   4096
#define HID     128
#define NT      8             // N tiles of 16
#define NWAVES  4
#define NTW     (NT / NWAVES) // 2 n-tiles per wave
#define KCH     5             // K chunks of 32 (160 = 128 + augment)
#define PW      160
#define PV8     (PW / 8)      // 20 v8h per h row

__device__ __forceinline__ float dev_tanh(float x) {
#if __has_builtin(__builtin_amdgcn_tanhf)
  return __builtin_amdgcn_tanhf(x);          // v_tanh_f32 TRANS op
#else
  float e = __expf(2.0f * x);
  return 1.0f - 2.0f * __builtin_amdgcn_rcpf(e + 1.0f);
#endif
}

#define WMMA(a, b, c) __builtin_amdgcn_wmma_f32_16x16x32_f16( \
    false, (a), false, (b), (short)0, (c), false, false)

__global__ void __launch_bounds__(32 * NWAVES)
rnn_wavefn_kernel(const float* __restrict__ data_in,  // [S,B,2] one-hot
                  const float* __restrict__ W_ih,     // [128,2]
                  const float* __restrict__ W_hh,     // [128,128]
                  const float* __restrict__ b_ih,     // [128]
                  const float* __restrict__ b_hh,     // [128]
                  const float* __restrict__ W_lin,    // [2,128]
                  const float* __restrict__ b_lin,    // [2]
                  float* __restrict__ out) {          // [S,B,2]
  __shared__ v16h  sB[NT * KCH * 32];   // staging: augmented W_hh^T fragments
  __shared__ v16h  sBlin[KCH * 32];     // staging: augmented W_lin^T fragments
  __shared__ v8h   sH[2][16 * PV8];     // double-buffered 16x160 f16 h tile
  __shared__ float sProb[32];

  const int tid   = threadIdx.x;
  const int wid   = tid >> 5;           // wave id 0..3
  const int lane  = tid & 31;
  const int col   = lane & 15;
  const int sel16 = (lane >> 4) << 4;   // 0|16 : B-fragment K half
  const int sel8  = (lane >> 4) << 3;   // 0|8  : A-fragment K half / D-row half
  const int b0    = blockIdx.x * 16;    // batch tile origin

  // ---- one-time init ----
  // K permutation: storage index p <-> hidden index j ; j = (p&7)*16 + (p>>3)
  for (int nl = 0; nl < NTW; ++nl) {
    int nt = wid * NTW + nl;
    for (int kc = 0; kc < KCH; ++kc) {
      int i = nt * 16 + col;            // B column = output hidden unit
      _Float16* dst = (_Float16*)&sB[(nt * KCH + kc) * 32 + lane];
      for (int e = 0; e < 16; ++e) {
        int p = kc * 32 + sel16 + e;
        float v;
        if (p < HID)       v = W_hh[i * HID + ((p & 7) * 16 + (p >> 3))];
        else if (p == 128) v = W_ih[i * 2 + 0] + b_ih[i] + b_hh[i];  // bias0
        else if (p == 129) v = W_ih[i * 2 + 1] + b_ih[i] + b_hh[i];  // bias1
        else               v = 0.0f;
        dst[e] = (_Float16)v;
      }
    }
  }
  if (wid == 0) {
    for (int kc = 0; kc < KCH; ++kc) {
      _Float16* dst = (_Float16*)&sBlin[kc * 32 + lane];
      for (int e = 0; e < 16; ++e) {
        int p = kc * 32 + sel16 + e;
        float v = 0.0f;
        if (col < 2) {
          if (p < HID)       v = W_lin[col * HID + ((p & 7) * 16 + (p >> 3))];
          else if (p == 130) v = b_lin[col];   // rides the constant-1 feature
        }
        dst[e] = (_Float16)v;
      }
    }
  }
  { // zero both h buffers (h0 = 0, augment tails = 0)
    v8h z = {};
    for (int q = tid; q < 2 * 16 * PV8; q += 32 * NWAVES) ((v8h*)sH)[q] = z;
  }
  __syncthreads();
  if (wid == 0 && lane < 16) {  // aug_0 = (x0=1, x1=0, 1, 0) into read buf of t=0
    uint2 aug;
    aug.x = 0x00003C00u;
    aug.y = 0x00003C00u;
    *(uint2*)&((_Float16*)sH[1])[lane * PW + HID] = aug;
  }
  __syncthreads();

  // ---- register-resident B fragments ----
  v16h Bfrag[NTW][KCH];
  v16h BlinR[KCH];
  #pragma unroll
  for (int nl = 0; nl < NTW; ++nl)
    #pragma unroll
    for (int kc = 0; kc < KCH; ++kc)
      Bfrag[nl][kc] = sB[((wid * NTW + nl) * KCH + kc) * 32 + lane];
  #pragma unroll
  for (int kc = 0; kc < KCH; ++kc) BlinR[kc] = sBlin[kc * 32 + lane];

  int cnt_up = 0;   // lanes 0..15: cum one-hot channel-0 for batch row b0+lane
  const v8f zf = {};

  // software-pipelined spin value: sv holds data_in[t] channel-1
  float sv = 0.0f;
  if (lane < 16) sv = data_in[(size_t)(b0 + lane) * 2 + 1];

  #pragma unroll 1
  for (int t = 0; t < S_SPINS; ++t) {
    const v8h* rb = sH[(t + 1) & 1];    // holds h_{t-1} + aug_t
    v8h*       wb = sH[t & 1];          // receives h_t + aug_{t+1}

    // ---- preload all 5 A chunks (distinct regs -> batched ds loads) ----
    v16h aa[KCH];
    #pragma unroll
    for (int kk = 0; kk < KCH; ++kk) {
      int ab = col * PV8 + kk * 4 + (sel8 >> 2);
      v8h alo = rb[ab];
      v8h ahi = rb[ab + 1];
      aa[kk] = __builtin_shufflevector(alo, ahi,
                 0,1,2,3,4,5,6,7,8,9,10,11,12,13,14,15);
    }

    // ---- recurrent GEMM: 4 partial chains, dep depth <= 3 ----
    v8f p0a = WMMA(aa[4], Bfrag[0][4], zf);   // aug chunk first (C = 0)
    v8f p1a = WMMA(aa[4], Bfrag[1][4], zf);
    v8f p0b = WMMA(aa[1], Bfrag[0][1], zf);
    v8f p1b = WMMA(aa[1], Bfrag[1][1], zf);
    p0a = WMMA(aa[0], Bfrag[0][0], p0a);
    p1a = WMMA(aa[0], Bfrag[1][0], p1a);
    p0b = WMMA(aa[2], Bfrag[0][2], p0b);
    p1b = WMMA(aa[2], Bfrag[1][2], p1b);
    p0b = WMMA(aa[3], Bfrag[0][3], p0b);
    p1b = WMMA(aa[3], Bfrag[1][3], p1b);
    v8f acc0 = p0a + p0b;
    v8f acc1 = p1a + p1b;

    // consume pipelined spin (loaded last iteration: covered by a full step)
    int s = (sv > 0.5f) ? 1 : 0;
    int cnt_mask = cnt_up;              // snapshot == cumsum through t-1
    cnt_up += 1 - s;                    // lanes >=16 accumulate junk, never read

    // h_t = tanh(acc) -> wb, 2 packed halves per row (one b32 store per r)
    #pragma unroll
    for (int r = 0; r < 8; ++r) {
      union { _Float16 h[2]; unsigned u; } pk;
      pk.h[0] = (_Float16)dev_tanh(acc0[r]);
      pk.h[1] = (_Float16)dev_tanh(acc1[r]);
      *(unsigned*)&((_Float16*)wb)[(sel8 + r) * PW + col * 8 + wid * NTW] = pk.u;
    }
    // aug_{t+1} -> wb (wave 0): p128=x0, p129=x1, p130=1, p131=0
    if (wid == 0 && lane < 16) {
      uint2 aug;
      aug.x = s ? 0x3C000000u : 0x00003C00u;
      aug.y = 0x00003C00u;
      *(uint2*)&((_Float16*)wb)[lane * PW + HID] = aug;
    }
    // issue next spin load + deep speculative prefetch (consumed next iter)
    if (lane < 16) {
      int tn = (t + 1 < S_SPINS) ? t + 1 : t;
      sv = data_in[((size_t)tn * BATCH + (b0 + lane)) * 2 + 1];
      if (wid == 0)
        __builtin_prefetch(&data_in[((size_t)(t + 16) * BATCH + (b0 + lane)) * 2], 0, 0);
    }
    __syncthreads();   // h_t complete & visible; protects sProb and buffers

    // ---- rotating logit duty: softmax + masking + store (5 WMMAs) ----
    // Reading wb's augment (aug_{t+1}) is safe: W_lin-frag spin rows are 0,
    // only the constant-1 row (b_lin) contributes.
    if (wid == (t & (NWAVES - 1))) {
      v16h la[KCH];
      #pragma unroll
      for (int kk = 0; kk < KCH; ++kk) {
        int ab = col * PV8 + kk * 4 + (sel8 >> 2);
        v8h alo = ((const v8h*)wb)[ab];
        v8h ahi = ((const v8h*)wb)[ab + 1];
        la[kk] = __builtin_shufflevector(alo, ahi,
                   0,1,2,3,4,5,6,7,8,9,10,11,12,13,14,15);
      }
      v8f LA = WMMA(la[4], BlinR[4], zf);
      v8f LB = WMMA(la[1], BlinR[1], zf);
      LA = WMMA(la[0], BlinR[0], LA);
      LB = WMMA(la[2], BlinR[2], LB);
      LB = WMMA(la[3], BlinR[3], LB);
      v8f accL = LA + LB;

      #pragma unroll
      for (int r = 0; r < 8; ++r) {
        float l  = accL[r];
        float lo = __shfl_xor(l, 1, 32);
        float p  = __builtin_amdgcn_rcpf(1.0f + __expf(lo - l));
        if (col < 2) sProb[(sel8 + r) * 2 + (lane & 1)] = p;
      }
      // masking at store time: one broadcast per storing lane
      float pv = sProb[lane];
      int cu = __shfl(cnt_mask, lane >> 1, 32);   // cum_up[t-1] of my row
      if (t > 0) {
        int cd = t - cu;                           // cum_down[t-1]
        if (cu >= S_SPINS / 2) pv = (lane & 1) ? 1.0f : 0.0f;  // down_guar
        if (cd >= S_SPINS / 2) pv = (lane & 1) ? 0.0f : 1.0f;  // up_guar
      }
      // coalesced 128B store: 16 rows x 2 probs
      out[((size_t)t * BATCH + b0) * 2 + lane] = pv;
    }
  }
}

extern "C" void kernel_launch(void* const* d_in, const int* in_sizes, int n_in,
                              void* d_out, int out_size, void* d_ws, size_t ws_size,
                              hipStream_t stream) {
  (void)in_sizes; (void)n_in; (void)out_size; (void)d_ws; (void)ws_size;
  const float* data_in = (const float*)d_in[0];
  const float* W_ih    = (const float*)d_in[1];
  const float* W_hh    = (const float*)d_in[2];
  const float* b_ih    = (const float*)d_in[3];
  const float* b_hh    = (const float*)d_in[4];
  const float* W_lin   = (const float*)d_in[5];
  const float* b_lin   = (const float*)d_in[6];
  float* out = (float*)d_out;

  rnn_wavefn_kernel<<<BATCH / 16, 32 * NWAVES, 0, stream>>>(
      data_in, W_ih, W_hh, b_ih, b_hh, W_lin, b_lin, out);
}
